// LDGuidedRetention_19662360281176
// MI455X (gfx1250) — compile-verified, matrix operand order checked by reading, not run on hardware
//
#include <hip/hip_runtime.h>
#include <math.h>
#include <stdint.h>

// ---------------------------------------------------------------------------
// LD-guided retention for MI455X (gfx1250, wave32, WMMA, async-LDS copies).
//   1) qkv  = x @ Wqkv + bqkv                  -> bf16 [B*L, 3D]     (ws)
//   2) S    = (q kT) * (1/sqrt(D)) * decaymask -> f32  [B, L, L]     (ws)
//   3) softmax rows of S, rewritten in place as bf16 P (row pitch 2L)
//   4) O    = P @ v                            -> bf16 [B*L, D]      (ws)
//   5) out  = O @ Wproj + bproj                -> f32  d_out
// ---------------------------------------------------------------------------

typedef __bf16 bf16;
typedef __attribute__((ext_vector_type(16))) __bf16 v16bf;
typedef __attribute__((ext_vector_type(8)))  float  v8f;
typedef unsigned int u32x4 __attribute__((ext_vector_type(4)));

#define BM 128
#define BN 128
#define BK 32
#define KP 40   // padded LDS row pitch in bf16 elems (80 B, keeps 16B alignment)

__device__ __forceinline__ void store_out(float* p, float v) { *p = v; }
__device__ __forceinline__ void store_out(bf16*  p, float v) { *p = (bf16)v; }

// 8-element load (+ f32->bf16 convert if needed) into a 16B register packet.
__device__ __forceinline__ void ld8(const float* g, bf16* o) {
  const float4* gp = (const float4*)g;
  float4 a = gp[0], b = gp[1];
  o[0] = (bf16)a.x; o[1] = (bf16)a.y; o[2] = (bf16)a.z; o[3] = (bf16)a.w;
  o[4] = (bf16)b.x; o[5] = (bf16)b.y; o[6] = (bf16)b.z; o[7] = (bf16)b.w;
}
__device__ __forceinline__ void ld8(const bf16* g, bf16* o) {
  *(u32x4*)o = *(const u32x4*)g;
}

// CDNA5 async global->LDS copy, 16 B per lane, tracked by ASYNCcnt.
__device__ __forceinline__ void async_cp16(uint32_t lds_off, const void* g) {
  asm volatile("global_load_async_to_lds_b128 %0, %1, off"
               :: "v"(lds_off), "v"((unsigned long long)g) : "memory");
}
__device__ __forceinline__ void async_fence() {
  asm volatile("s_wait_asynccnt 0x0" ::: "memory");
}

// 16-bit A/B fragment (16x32) from an LDS tile stored [row][k] with pitch KP.
// Lane L: row = L%16; halfs 0..7 hold K=(L/16)*8..+7, halfs 8..15 hold
// K=16+(L/16)*8..+7 (CDNA5 ISA 7.12.2 16-bit A layout).
__device__ __forceinline__ v16bf ld_frag(const bf16* tile, int lane) {
  const int r = lane & 15, h = lane >> 4;
  const bf16* p = tile + r * KP + h * 8;
  union { v16bf v; u32x4 q[2]; } u;
  u.q[0] = *(const u32x4*)(p);
  u.q[1] = *(const u32x4*)(p + 16);
  return u.v;
}

enum { EPI_BIAS = 0, EPI_MASK = 1 };

template <typename TA, typename TB, typename TC, bool TRANS_B, int EPI>
__global__ __launch_bounds__(256) void gemm_wmma(
    const TA* __restrict__ A, long lda, long sAb,
    const TB* __restrict__ B, long ldb, long sBb,
    TC* __restrict__ C, long ldc, long sCb,
    const float* __restrict__ bias,
    int M, int N, int K,
    const float* __restrict__ gammap, float scale, int window)
{
  constexpr bool A_ASY = (sizeof(TA) == 2);               // bf16 A -> async copy
  constexpr bool B_ASY = (sizeof(TB) == 2) && TRANS_B;    // bf16 row-major B^T

  // double-buffered LDS: [buf][As rows 0..127 | Bs rows 128..255][KP]
  __shared__ alignas(16) bf16 smem[2][BM + BN][KP];

  const int tid  = threadIdx.x;
  const int lane = tid & 31;
  const int wid  = tid >> 5;
  const int wm   = wid >> 2;   // 0..1 : 64 rows per wave
  const int wn   = wid & 3;    // 0..3 : 32 cols per wave

  const long zb = blockIdx.z;
  const TA* Ab = A + zb * sAb;
  const TB* Bb = B + zb * sBb;
  TC*       Cb = C + zb * sCb;

  const int row0 = blockIdx.y * BM;
  const int col0 = blockIdx.x * BN;

  const uint32_t smem0    = (uint32_t)(uintptr_t)&smem[0][0][0];
  const uint32_t bufBytes = (BM + BN) * KP * sizeof(bf16);

  // --- per-thread staging coords (precomputed once; advanced per chunk) ----
  const TA* pA[2]; uint32_t lA[2];            // A: 2 x 8-elem packets/thread
  #pragma unroll
  for (int e = 0; e < 2; ++e) {
    int v = tid + 256 * e;
    int r = v >> 2, c8 = (v & 3) * 8;
    pA[e] = Ab + (long)(row0 + r) * lda + c8;
    lA[e] = (uint32_t)(r * KP + c8);
  }
  const TB* pB[2]; uint32_t lB[2];
  #pragma unroll
  for (int e = 0; e < 2; ++e) {
    int v = tid + 256 * e;
    if (TRANS_B) {             // global B is [N][K]: row-major into Bs[n][k]
      int r = v >> 2, c8 = (v & 3) * 8;
      pB[e] = Bb + (long)(col0 + r) * ldb + c8;
      lB[e] = (uint32_t)((BM + r) * KP + c8);
    } else {                   // global B is [K][N]: transpose into Bs[n][k]
      int k = v >> 4, n8 = (v & 15) * 8;
      pB[e] = Bb + (long)k * ldb + col0 + n8;
      lB[e] = (uint32_t)((BM + n8) * KP + k);
    }
  }

  auto stageA = [&](int buf) {
    #pragma unroll
    for (int e = 0; e < 2; ++e) {
      bf16* dst = &smem[buf][0][0] + lA[e];
      if constexpr (A_ASY) {
        async_cp16(smem0 + buf * bufBytes + lA[e] * 2, pA[e]);
      } else {
        union { bf16 h[8]; u32x4 q; } t;
        ld8(pA[e], t.h);
        *(u32x4*)dst = t.q;
      }
      pA[e] += BK;
    }
  };
  auto stageB = [&](int buf) {
    #pragma unroll
    for (int e = 0; e < 2; ++e) {
      if constexpr (TRANS_B) {
        bf16* dst = &smem[buf][0][0] + lB[e];
        if constexpr (B_ASY) {
          async_cp16(smem0 + buf * bufBytes + lB[e] * 2, pB[e]);
        } else {
          union { bf16 h[8]; u32x4 q; } t;
          ld8(pB[e], t.h);
          *(u32x4*)dst = t.q;
        }
        pB[e] += BK;
      } else {
        union { bf16 h[8]; u32x4 q; } t;
        ld8(pB[e], t.h);
        bf16* base = &smem[buf][0][0] + lB[e];
        #pragma unroll
        for (int j = 0; j < 8; ++j) base[j * KP] = t.h[j];  // scatter along n
        pB[e] += (long)BK * ldb;
      }
    }
  };

  v8f acc[4][2] = {};
  // Preload all 6 fragments, then 8 back-to-back WMMAs: one ds_load clause,
  // partial s_wait_dscnt, XDL pipe stays fed.
  auto compute = [&](int buf) {
    const bf16* base = &smem[buf][0][0];
    v16bf b0 = ld_frag(base + (BM + wn * 32 +  0) * KP, lane);
    v16bf b1 = ld_frag(base + (BM + wn * 32 + 16) * KP, lane);
    v16bf a0 = ld_frag(base + (wm * 64 +  0) * KP, lane);
    v16bf a1 = ld_frag(base + (wm * 64 + 16) * KP, lane);
    v16bf a2 = ld_frag(base + (wm * 64 + 32) * KP, lane);
    v16bf a3 = ld_frag(base + (wm * 64 + 48) * KP, lane);
    acc[0][0] = __builtin_amdgcn_wmma_f32_16x16x32_bf16(false, a0, false, b0, (short)0, acc[0][0], false, false);
    acc[0][1] = __builtin_amdgcn_wmma_f32_16x16x32_bf16(false, a0, false, b1, (short)0, acc[0][1], false, false);
    acc[1][0] = __builtin_amdgcn_wmma_f32_16x16x32_bf16(false, a1, false, b0, (short)0, acc[1][0], false, false);
    acc[1][1] = __builtin_amdgcn_wmma_f32_16x16x32_bf16(false, a1, false, b1, (short)0, acc[1][1], false, false);
    acc[2][0] = __builtin_amdgcn_wmma_f32_16x16x32_bf16(false, a2, false, b0, (short)0, acc[2][0], false, false);
    acc[2][1] = __builtin_amdgcn_wmma_f32_16x16x32_bf16(false, a2, false, b1, (short)0, acc[2][1], false, false);
    acc[3][0] = __builtin_amdgcn_wmma_f32_16x16x32_bf16(false, a3, false, b0, (short)0, acc[3][0], false, false);
    acc[3][1] = __builtin_amdgcn_wmma_f32_16x16x32_bf16(false, a3, false, b1, (short)0, acc[3][1], false, false);
  };

  const int nchunk = K / BK;
  stageA(0); stageB(0);
  int buf = 0;
  for (int ck = 0; ck < nchunk; ++ck) {
    if constexpr (A_ASY || B_ASY) async_fence();
    __syncthreads();
    if (ck + 1 < nchunk) { stageA(buf ^ 1); stageB(buf ^ 1); }
    compute(buf);
    buf ^= 1;
  }

  float lg = 0.f;
  if (EPI == EPI_MASK) lg = __log2f(gammap[0]);

  // C/D 16x16 f32 layout: m = e + 8*(lane>>4), n = lane&15
  #pragma unroll
  for (int mt = 0; mt < 4; ++mt) {
    #pragma unroll
    for (int nt = 0; nt < 2; ++nt) {
      #pragma unroll
      for (int e = 0; e < 8; ++e) {
        int m = row0 + wm * 64 + mt * 16 + ((lane >> 4) * 8) + e;
        int n = col0 + wn * 32 + nt * 16 + (lane & 15);
        float v = acc[mt][nt][e];
        if (EPI == EPI_MASK) {
          v *= scale;
          if ((m / window) == (n / window)) {
            int d = m - n;
            v = (d >= 0) ? v * exp2f((float)d * lg) : 0.f;
          }
        } else {
          if (bias) v += bias[n];
        }
        store_out(&Cb[(long)m * ldc + n], v);
      }
    }
  }
}

// One block per row: f32 softmax over L, rewrite row in place as bf16
// (P occupies the first 2*L bytes of each 4*L-byte row -> row pitch 2L bf16).
__global__ __launch_bounds__(256) void softmax_row_bf16(float* __restrict__ S, int L) {
  const long row = blockIdx.x;
  float* rp = S + row * (long)L;
  bf16*  op = (bf16*)rp;
  const int tid = threadIdx.x;
  const int per = L / 256;               // 16 for L = 4096
  float vals[16];
  float mx = -3.4e38f;
  for (int e = 0; e < per; ++e) {
    vals[e] = rp[tid + 256 * e];
    mx = fmaxf(mx, vals[e]);
  }
  __shared__ float red[256];
  red[tid] = mx; __syncthreads();
  for (int s = 128; s > 0; s >>= 1) {
    if (tid < s) red[tid] = fmaxf(red[tid], red[tid + s]);
    __syncthreads();
  }
  mx = red[0]; __syncthreads();
  float sm = 0.f;
  for (int e = 0; e < per; ++e) { vals[e] = __expf(vals[e] - mx); sm += vals[e]; }
  red[tid] = sm; __syncthreads();
  for (int s = 128; s > 0; s >>= 1) {
    if (tid < s) red[tid] += red[tid + s];
    __syncthreads();
  }
  const float inv = 1.f / red[0]; __syncthreads();
  for (int e = 0; e < per; ++e)
    op[tid + 256 * e] = (bf16)(vals[e] * inv);
}

extern "C" void kernel_launch(void* const* d_in, const int* in_sizes, int n_in,
                              void* d_out, int out_size, void* d_ws, size_t ws_size,
                              hipStream_t stream) {
  const float* x     = (const float*)d_in[0];
  const float* Wqkv  = (const float*)d_in[1];
  const float* bqkv  = (const float*)d_in[2];
  const float* Wproj = (const float*)d_in[3];
  const float* bproj = (const float*)d_in[4];
  const float* gamma = (const float*)d_in[5];
  float* out = (float*)d_out;

  const int D = 1024, W = 128, B = 2;
  const long BL = (long)in_sizes[0] / D;   // B*L = 8192
  const int  L  = (int)(BL / B);           // 4096
  const float scale = 1.0f / sqrtf((float)D);

  // workspace layout (~201 MB): qkv bf16 | S f32 | O bf16
  char* wsb = (char*)d_ws;
  size_t off = 0;
  bf16* qkv = (bf16*)(wsb + off);
  off += ((size_t)BL * 3 * D * sizeof(bf16) + 255) & ~(size_t)255;
  float* S = (float*)(wsb + off);
  off += ((size_t)B * (size_t)L * (size_t)L * sizeof(float) + 255) & ~(size_t)255;
  bf16* O = (bf16*)(wsb + off);

  // 1) qkv = x @ Wqkv + bqkv
  {
    dim3 g(3 * D / BN, (unsigned)(BL / BM), 1);
    gemm_wmma<float, float, bf16, false, EPI_BIAS><<<g, 256, 0, stream>>>(
        x, D, 0, Wqkv, 3 * D, 0, qkv, 3 * D, 0,
        bqkv, (int)BL, 3 * D, D, nullptr, 1.f, W);
  }
  // 2) S = (q kT) * scale * decay-mask  (per batch; both operands async bf16)
  {
    dim3 g(L / BN, L / BM, B);
    gemm_wmma<bf16, bf16, float, true, EPI_MASK><<<g, 256, 0, stream>>>(
        qkv, 3 * D, (long)L * 3 * D, qkv + D, 3 * D, (long)L * 3 * D,
        S, L, (long)L * L, nullptr, L, L, D, gamma, scale, W);
  }
  // 3) row softmax; P written in place as bf16 with row pitch 2L
  softmax_row_bf16<<<dim3((unsigned)(B * L)), 256, 0, stream>>>(S, L);
  // 4) O = P @ v   (A = P async bf16)
  {
    dim3 g(D / BN, L / BM, B);
    gemm_wmma<bf16, bf16, bf16, false, EPI_BIAS><<<g, 256, 0, stream>>>(
        (const bf16*)S, 2 * (long)L, (long)L * 2 * (long)L,
        qkv + 2 * D, 3 * D, (long)L * 3 * D,
        O, D, (long)L * D, nullptr, L, D, L, nullptr, 1.f, W);
  }
  // 5) out = O @ Wproj + bproj
  {
    dim3 g(D / BN, (unsigned)(BL / BM), 1);
    gemm_wmma<bf16, float, float, false, EPI_BIAS><<<g, 256, 0, stream>>>(
        O, D, 0, Wproj, D, 0, out, D, 0,
        bproj, (int)BL, D, D, nullptr, 1.f, W);
  }
}
